// MEWISPool_69896297775674
// MI455X (gfx1250) — compile-verified, matrix operand order checked by reading, not run on hardware
//
#include <hip/hip_runtime.h>
#include <cstdint>

// ---------------------------------------------------------------------------
// MEWISPool for MI455X (gfx1250, wave32, WMMA + TDM).
//  - A2=A@A, A3=A2@A      : bf16 WMMA 16x16x32 (exact: operands are small ints)
//                           with TDM (tensor_load_to_lds) double-buffered B tiles
//  - A@[x,x^2], A@h, MLPs : f32  WMMA 16x16x4  (exact fp32 path)
//  - greedy MIS scan      : incremental rank-update, 1 WG, LDS atomics
// ---------------------------------------------------------------------------

constexpr int kN   = 2048;
constexpr int kD   = 256;
constexpr int kHid = 128;
constexpr int kG   = 8;
constexpr int kE2  = 16384;           // directed edges = 2*kE2
constexpr int kNN  = kN * kN;

typedef __attribute__((ext_vector_type(16))) __bf16 v16bf;
typedef __attribute__((ext_vector_type(8)))  float  v8f;
typedef __attribute__((ext_vector_type(2)))  float  v2f;

// ------------------------- Tensor Data Mover hooks -------------------------

#if defined(__HIP_DEVICE_COMPILE__) && \
    __has_builtin(__builtin_amdgcn_tensor_load_to_lds) && \
    __has_builtin(__builtin_amdgcn_s_wait_tensorcnt)
#define USE_TDM 1
#else
#define USE_TDM 0
#endif

#if USE_TDM
typedef __attribute__((ext_vector_type(4))) unsigned tdm_u32x4;
typedef __attribute__((ext_vector_type(8))) int      tdm_i32x8;
typedef __attribute__((ext_vector_type(4))) int      tdm_i32x4;

// Issue one TDM 2-D tile load: 32 rows x 16 cols of bf16 (2 B elements),
// row stride = row_stride elements, into LDS byte offset lds_off.
// D# layout per CDNA5 ISA ch.8 (group0: count/lds/global/type, group1: dims).
__device__ __forceinline__ void tdm_load_bf16_tile(const __bf16* gptr,
                                                   unsigned lds_off,
                                                   int row_stride,
                                                   int tensor_rows) {
    unsigned long long ga = (unsigned long long)(uintptr_t)gptr;
    tdm_u32x4 g0 = {
        1u,                                                  // count=1 (valid), user
        lds_off,                                             // D#.lds_addr (bytes)
        (unsigned)ga,                                        // global_addr[31:0]
        (unsigned)((ga >> 32) & 0x01ffffffu) | (2u << 30)    // addr[56:32] | type=2
    };
    unsigned td0 = (unsigned)row_stride;                     // tensor_dim0 (elems/row)
    unsigned td1 = (unsigned)tensor_rows;                    // tensor_dim1 (rows)
    tdm_i32x8 g1 = {
        (int)0x00010000u,                                    // data_size=1 (2 bytes)
        (int)((td0 & 0xffffu) << 16),                        // abar_addr=0 | dim0 lo16
        (int)(((td0 >> 16) & 0xffffu) | ((td1 & 0xffffu) << 16)),
        (int)(((td1 >> 16) & 0xffffu) | (16u << 16)),        // tile_dim0 = 16
        32,                                                  // tile_dim1 = 32
        (int)td0,                                            // dim0_stride lo32
        0, 0                                                 // dim0_stride hi / dim1_stride
    };
    tdm_i32x4 gz = {0, 0, 0, 0};
#if defined(__clang_major__) && (__clang_major__ >= 23)
    tdm_i32x8 z8 = {0, 0, 0, 0, 0, 0, 0, 0};
    __builtin_amdgcn_tensor_load_to_lds(g0, g1, gz, gz, z8, 0);
#else
    __builtin_amdgcn_tensor_load_to_lds(g0, g1, gz, gz, 0);
#endif
}
#endif

// ------------------------------ small helpers ------------------------------

__device__ __forceinline__ float blockReduceSum(float v, float* red) {
    int t = threadIdx.x;
    red[t] = v; __syncthreads();
    for (int s = blockDim.x >> 1; s > 0; s >>= 1) {
        if (t < s) red[t] += red[t + s];
        __syncthreads();
    }
    float r = red[0]; __syncthreads();
    return r;
}

__device__ __forceinline__ float blockReduceMax(float v, float* red) {
    int t = threadIdx.x;
    red[t] = v; __syncthreads();
    for (int s = blockDim.x >> 1; s > 0; s >>= 1) {
        if (t < s) red[t] = fmaxf(red[t], red[t + s]);
        __syncthreads();
    }
    float r = red[0]; __syncthreads();
    return r;
}

__global__ void k_zero(float* p, int n) {
    int i = blockIdx.x * blockDim.x + threadIdx.x;
    if (i < n) p[i] = 0.f;
}
__global__ void k_zeroi(int* p, int n) {
    int i = blockIdx.x * blockDim.x + threadIdx.x;
    if (i < n) p[i] = 0;
}
__global__ void k_copy(const float* s, float* d, int n) {
    int i = blockIdx.x * blockDim.x + threadIdx.x;
    if (i < n) d[i] = s[i];
}
__global__ void k_f2bf(const float* s, __bf16* d, int n) {
    int i = blockIdx.x * blockDim.x + threadIdx.x;
    if (i < n) d[i] = (__bf16)s[i];
}
__global__ void k_add_inplace(float* dst, const float* src, int n) {
    int i = blockIdx.x * blockDim.x + threadIdx.x;
    if (i < n) dst[i] += src[i];
}

// ------------------------------ graph build --------------------------------

__global__ void k_scatter_edges(const int* ei, float* A) {
    int t = blockIdx.x * blockDim.x + threadIdx.x;
    if (t < 2 * kE2) {
        int r = ei[t];
        int c = ei[2 * kE2 + t];
        A[r * kN + c] = 1.f;           // duplicates write the same value: OK
    }
}

__global__ void k_rowsum(const float* __restrict__ A, float* deg) {
    __shared__ float red[256];
    int row = blockIdx.x;
    float s = 0.f;
    for (int j = threadIdx.x; j < kN; j += blockDim.x) s += A[row * kN + j];
    s = blockReduceSum(s, red);
    if (threadIdx.x == 0) deg[row] = s;
}

__global__ void k_make_bcat(const float* __restrict__ x, float* __restrict__ B) {
    int i = blockIdx.x * blockDim.x + threadIdx.x;   // over N*D
    if (i < kN * kD) {
        int r = i / kD, c = i % kD;
        float v = x[i];
        B[r * (2 * kD) + c]       = v;
        B[r * (2 * kD) + kD + c]  = v * v;
    }
}

// ------------------------- WMMA GEMMs (CDNA5) ------------------------------
// C[M,Nc] = A[M,K] @ B[K,Nc], all row-major fp32. One wave per 16x16 C tile.
// Fragment layouts per CDNA5 ISA 7.12.2 (wave32):
//   A 16x4 f32 : lane l holds M=l%16; K = (l<16?0:2)+{0,1}
//   C   16x16  : VGPR r holds M = r + (l<16?0:8), N = l%16
__global__ void k_gemm_f32(const float* __restrict__ A, const float* __restrict__ B,
                           float* __restrict__ C, int M, int K, int Nc) {
    int wave = threadIdx.x >> 5;
    int lane = threadIdx.x & 31;
    int tilesN = Nc >> 4;
    int tile = blockIdx.x * 8 + wave;
    if (tile >= (M >> 4) * tilesN) return;
    int tm = (tile / tilesN) << 4;
    int tn = (tile % tilesN) << 4;
    int l16 = lane & 15;
    int mrow = tm + l16;
    int ncol = tn + l16;
    int kh = (lane < 16) ? 0 : 2;

    v8f c = {};
    for (int kb = 0; kb < K; kb += 4) {
        if ((kb & 63) == 0 && kb + 256 < K)   // pull next A rows toward L0
            __builtin_prefetch(&A[mrow * K + kb + 256], 0, 1);
        v2f a, b;
        a.x = A[mrow * K + kb + kh];
        a.y = A[mrow * K + kb + kh + 1];
        b.x = B[(kb + kh) * Nc + ncol];
        b.y = B[(kb + kh + 1) * Nc + ncol];
        c = __builtin_amdgcn_wmma_f32_16x16x4_f32(false, a, false, b,
                                                  (short)0, c, false, false);
    }
    int mo = tm + (lane < 16 ? 0 : 8);
#pragma unroll
    for (int r = 0; r < 8; ++r)
        C[(mo + r) * Nc + ncol] = c[r];
}

// bf16 WMMA GEMM: block = 8 waves computes a 128(M) x 16(N) slab. The 32x16
// B chunk is DMA'd into a double-buffered LDS slab by the Tensor Data Mover
// (wave 0 drives the D#; s_wait_tensorcnt + barrier publish it), overlapping
// the DMA of tile k+1 with WMMA on tile k. Fallback: cooperative VALU stage.
// A 16x32 bf16 layout: lane l holds M=l%16; v[0..7] K=kb+{0..7}+(l<16?0:8),
// v[8..15] K=kb+16+{0..7}+(l<16?0:8).  B mirrors with N=l%16.
__global__ void k_gemm_bf16(const __bf16* __restrict__ A, const __bf16* __restrict__ B,
                            float* __restrict__ C, int M, int K, int Nc) {
    __shared__ __bf16 ldsB[2][32 * 16];    // first (only) LDS object -> offset 0
    int wave = threadIdx.x >> 5;
    int lane = threadIdx.x & 31;
    int l16 = lane & 15;
    int tm = blockIdx.y * 128 + wave * 16;
    int tn = blockIdx.x * 16;
    int kbase = (lane < 16) ? 0 : 8;
    int mrow = tm + l16;

    v8f c = {};

#if USE_TDM
    if (wave == 0) {                       // prologue: DMA tile 0 into buffer 0
        tdm_load_bf16_tile(B + tn, 0u, Nc, K);
        __builtin_amdgcn_s_wait_tensorcnt(0);
    }
    __syncthreads();
#endif

    for (int kb = 0; kb < K; kb += 32) {
#if USE_TDM
        const int cur = (kb >> 5) & 1;
        const bool more = (kb + 32) < K;
        if (wave == 0 && more)             // overlap: DMA next tile into other buf
            tdm_load_bf16_tile(B + (kb + 32) * Nc + tn,
                               (unsigned)((cur ^ 1) * 32 * 16 * 2), Nc, K);
#else
        const int cur = 0;
        {   // cooperative stage of B[kb..kb+32, tn..tn+16]
            int e = threadIdx.x;           // 256 threads x 2 elems = 512
            int r0 = e >> 3, c0 = (e & 7) * 2;
            ldsB[0][r0 * 16 + c0]     = B[(kb + r0) * Nc + tn + c0];
            ldsB[0][r0 * 16 + c0 + 1] = B[(kb + r0) * Nc + tn + c0 + 1];
        }
        __syncthreads();
#endif
        v16bf a, b;
#pragma unroll
        for (int i = 0; i < 4; ++i) {
            a[2 * i]         = A[mrow * K + kb + kbase + 2 * i];
            a[2 * i + 1]     = A[mrow * K + kb + kbase + 2 * i + 1];
            a[8 + 2 * i]     = A[mrow * K + kb + kbase + 16 + 2 * i];
            a[8 + 2 * i + 1] = A[mrow * K + kb + kbase + 16 + 2 * i + 1];
            b[2 * i]         = ldsB[cur][(kbase + 2 * i) * 16 + l16];
            b[2 * i + 1]     = ldsB[cur][(kbase + 2 * i + 1) * 16 + l16];
            b[8 + 2 * i]     = ldsB[cur][(kbase + 16 + 2 * i) * 16 + l16];
            b[8 + 2 * i + 1] = ldsB[cur][(kbase + 16 + 2 * i + 1) * 16 + l16];
        }
        c = __builtin_amdgcn_wmma_f32_16x16x32_bf16(false, a, false, b,
                                                    (short)0, c, false, false);
#if USE_TDM
        if (wave == 0 && more)
            __builtin_amdgcn_s_wait_tensorcnt(0);   // next buffer landed
        __syncthreads();
#else
        __syncthreads();
#endif
    }
    int mo = tm + (lane < 16 ? 0 : 8);
#pragma unroll
    for (int r = 0; r < 8; ++r)
        C[(mo + r) * Nc + tn + l16] = c[r];
}

// row-wise dot:  out[row] = sum_j A[row, 0..nc) * v[j]
__global__ void k_gemv(const float* __restrict__ A, const float* __restrict__ v,
                       float* out, int nc) {
    __shared__ float red[256];
    int row = blockIdx.x;
    float s = 0.f;
    for (int j = threadIdx.x; j < nc; j += blockDim.x) s += A[row * nc + j] * v[j];
    s = blockReduceSum(s, red);
    if (threadIdx.x == 0) out[row] = s;
}

// ------------------------------ entropy path -------------------------------

__global__ void k_compute_V(const float* __restrict__ x, const float* __restrict__ AxAxx,
                            const float* __restrict__ deg, float* V) {
    __shared__ float red[256];
    int i = blockIdx.x;
    float dg = deg[i];
    float s = 0.f;
    for (int c = threadIdx.x; c < kD; c += blockDim.x) {
        float xv  = x[i * kD + c];
        float ax  = AxAxx[i * 2 * kD + c];
        float axx = AxAxx[i * 2 * kD + kD + c];
        float v = dg * xv * xv - 2.f * xv * ax + axx;   // x*(deg*x-Ax) - x*Ax + A(x*x)
        s += v * v;
    }
    s = blockReduceSum(s, red);
    if (threadIdx.x == 0) V[i] = sqrtf(s);
}

__global__ void k_segmax(const float* V, const int* batch, float* segmx) {
    __shared__ float red[256];
    int g = blockIdx.x;
    float m = -3.4e38f;
    for (int i = threadIdx.x; i < kN; i += blockDim.x)
        if (batch[i] == g) m = fmaxf(m, V[i]);
    m = blockReduceMax(m, red);
    if (threadIdx.x == 0) segmx[g] = m;
}

__global__ void k_segsumexp(const float* V, const int* batch, const float* segmx,
                            float* segsm) {
    __shared__ float red[256];
    int g = blockIdx.x;
    float mx = segmx[g];
    float s = 0.f;
    for (int i = threadIdx.x; i < kN; i += blockDim.x)
        if (batch[i] == g) s += expf(V[i] - mx);
    s = blockReduceSum(s, red);
    if (threadIdx.x == 0) segsm[g] = s;
}

__global__ void k_ent(const float* V, const int* batch, const float* segmx,
                      const float* segsm, float* ent, float* gamma) {
    __shared__ float red[256];
    int i = blockIdx.x * blockDim.x + threadIdx.x;
    float e = 0.f;
    if (i < kN) {
        int g = batch[i];
        float P = expf(V[i] - segmx[g]) / segsm[g];
        if (P == 0.f) P = 1.f;
        e = -P * logf(P);
        ent[i] = e;
    }
    e = blockReduceSum(e, red);
    if (threadIdx.x == 0) atomicAdd(gamma, e);
}

// ------------------------------ MLP pieces ---------------------------------

__global__ void k_outer_lin1(const float* agg1, const float* w, const float* b,
                             float* z) {
    int i = blockIdx.x * blockDim.x + threadIdx.x;
    if (i < kN * kHid) {
        int r = i / kHid, c = i % kHid;
        z[i] = fmaxf(agg1[r] * w[c] + b[c], 0.f);
    }
}
__global__ void k_bias_relu(float* z, const float* b, int n, int cols) {
    int i = blockIdx.x * blockDim.x + threadIdx.x;
    if (i < n) z[i] = fmaxf(z[i] + b[i % cols], 0.f);
}
__global__ void k_bias(float* z, const float* b, int n, int cols) {
    int i = blockIdx.x * blockDim.x + threadIdx.x;
    if (i < n) z[i] += b[i % cols];
}
__global__ void k_bnstats(const float* z, float* mu, float* var) {
    __shared__ float rs[256];
    __shared__ float rq[256];
    int c = blockIdx.x;
    float s = 0.f, q = 0.f;
    for (int r = threadIdx.x; r < kN; r += blockDim.x) {
        float v = z[r * kHid + c];
        s += v; q += v * v;
    }
    s = blockReduceSum(s, rs);
    q = blockReduceSum(q, rq);
    if (threadIdx.x == 0) {
        float m = s / (float)kN;
        mu[c] = m;
        var[c] = q / (float)kN - m * m;   // biased batch variance
    }
}
__global__ void k_bnapply(float* z, const float* mu, const float* var,
                          const float* g, const float* be) {
    int i = blockIdx.x * blockDim.x + threadIdx.x;
    if (i < kN * kHid) {
        int c = i % kHid;
        z[i] = (z[i] - mu[c]) * rsqrtf(var[c] + 1e-5f) * g[c] + be[c];
    }
}
__global__ void k_sigmoid_bias(const float* logits, const float* b, float* p) {
    int i = blockIdx.x * blockDim.x + threadIdx.x;
    if (i < kN) p[i] = 1.f / (1.f + expf(-(logits[i] + b[0])));
}

// ------------------------------- loss --------------------------------------

__global__ void k_dot(const float* a, const float* b, float* out, int n) {
    __shared__ float red[256];
    int i = blockIdx.x * blockDim.x + threadIdx.x;
    float v = (i < n) ? a[i] * b[i] : 0.f;
    v = blockReduceSum(v, red);
    if (threadIdx.x == 0) atomicAdd(out, v);
}
// scal: [0]=gamma [1]=loss [2]=ent.p [3]=p.Ap [4]=e_dot [5]=quad
__global__ void k_finalize_loss(float* scal, float* out_loss) {
    float L = scal[0] - scal[2] + scal[3];
    scal[1] = L;
    scal[4] = scal[2];
    scal[5] = scal[3];
    *out_loss = L;
}

// --------------------------- sort (bitonic, desc) --------------------------

__global__ void k_sort(const float* p, int* order) {
    __shared__ float key[kN];
    __shared__ int   val[kN];
    for (int i = threadIdx.x; i < kN; i += blockDim.x) { key[i] = p[i]; val[i] = i; }
    __syncthreads();
    for (int k = 2; k <= kN; k <<= 1) {
        for (int j = k >> 1; j > 0; j >>= 1) {
            for (int i = threadIdx.x; i < kN; i += blockDim.x) {
                int ixj = i ^ j;
                if (ixj > i) {
                    bool desc = ((i & k) == 0);
                    bool sw = desc ? (key[i] < key[ixj]) : (key[i] > key[ixj]);
                    if (sw) {
                        float tk = key[i]; key[i] = key[ixj]; key[ixj] = tk;
                        int   tv = val[i]; val[i] = val[ixj]; val[ixj] = tv;
                    }
                }
            }
            __syncthreads();
        }
    }
    for (int i = threadIdx.x; i < kN; i += blockDim.x) order[i] = val[i];
}

// --------------------- greedy scan (incremental, 1 WG) ---------------------
// State:  dummy = current s;  q = A @ dummy;  e_dot = ent.dummy;
//         quad = dummy^T A dummy.  Candidate s differs from dummy only on
//         supp = {neighbors with dummy!=0} U {idx}; evaluate li via deltas.
__global__ void k_greedy(const float* __restrict__ A, const float* __restrict__ ent,
                         const int* __restrict__ order, const float* scal,
                         float* dummy, float* q, int* sel, int* rej) {
    __shared__ int   s_idx, s_cnt, s_any, s_accept;
    __shared__ float s_dent, s_dq, s_cross, s_edot, s_quad;
    __shared__ int   suppJ[kN];
    __shared__ float suppD[kN];

    const float gamma = scal[0];
    const float loss  = scal[1];
    const int tid = threadIdx.x, NT = blockDim.x;
    if (tid == 0) { s_edot = scal[4]; s_quad = scal[5]; }
    __syncthreads();

    for (int step = 0; step < kN; ++step) {
        if (tid == 0) {
            s_idx = order[step];
            s_cnt = 0; s_any = 0; s_accept = 0;
            s_dent = 0.f; s_dq = 0.f; s_cross = 0.f;
        }
        __syncthreads();
        const int idx = s_idx;
        const float* rowA = A + idx * kN;

        float pe = 0.f, pq = 0.f;
        for (int j = tid; j < kN; j += NT) {
            if (rowA[j] > 0.f) {
                s_any = 1;                                  // benign race
                float dj = -dummy[j];
                if (dj != 0.f) {
                    int slot = atomicAdd(&s_cnt, 1);
                    suppJ[slot] = j; suppD[slot] = dj;
                    pe += dj * ent[j];
                    pq += dj * q[j];
                }
            }
        }
        if (tid == 0) {                                     // idx term: s[idx]=1
            float di = 1.f - dummy[idx];
            if (di != 0.f) {
                int slot = atomicAdd(&s_cnt, 1);
                suppJ[slot] = idx; suppD[slot] = di;
                pe += di * ent[idx];
                pq += di * q[idx];
            }
        }
        atomicAdd(&s_dent, pe);                             // ds_add_f32
        atomicAdd(&s_dq, pq);
        __syncthreads();

        const int S = s_cnt;
        float pc = 0.f;
        for (int pidx = tid; pidx < S * S; pidx += NT) {    // delta^T A delta
            int a_ = pidx / S, b_ = pidx % S;
            pc += suppD[a_] * A[suppJ[a_] * kN + suppJ[b_]] * suppD[b_];
        }
        atomicAdd(&s_cross, pc);
        __syncthreads();

        if (tid == 0) {
            bool deg0 = (s_any == 0);
            int selv = sel[idx] | (deg0 ? 1 : 0);
            sel[idx] = selv;
            bool eligible = (!rej[idx]) && (!selv);
            float li = gamma - (s_edot + s_dent) + (s_quad + 2.f * s_dq + s_cross);
            if (eligible && li <= loss) {
                s_accept = 1;
                sel[idx] = 1;
                s_edot += s_dent;
                s_quad += 2.f * s_dq + s_cross;
            }
        }
        __syncthreads();

        if (s_accept) {
            for (int t = tid; t < S; t += NT) dummy[suppJ[t]] += suppD[t];
            for (int j = tid; j < kN; j += NT)
                if (rowA[j] > 0.f) rej[j] = 1;              // rej |= nb
            for (int k = tid; k < kN; k += NT) {            // q += A @ delta (A sym)
                float acc = q[k];
                for (int t = 0; t < S; ++t)
                    acc += suppD[t] * A[suppJ[t] * kN + k];
                q[k] = acc;
            }
        }
        __syncthreads();
    }
}

// ------------------------------- outputs -----------------------------------

__global__ void k_outputs_node(const float* x, const int* sel, const int* batch,
                               float* out_x, float* out_sel, float* out_bp) {
    int i = blockIdx.x * blockDim.x + threadIdx.x;
    if (i < kN * kD) {
        int r = i / kD;
        out_x[i] = x[i] * (sel[r] ? 1.f : 0.f);
    }
    if (i < kN) {
        out_sel[i] = sel[i] ? 1.f : 0.f;
        out_bp[i]  = sel[i] ? (float)batch[i] : -1.f;
    }
}
__global__ void k_adj_final(const float* A2, float* out_adj, const int* sel) {
    int i = blockIdx.x * blockDim.x + threadIdx.x;
    if (i >= kNN) return;
    int r = i / kN, c = i % kN;
    float v = fminf(fmaxf(A2[i] + out_adj[i], 0.f), 1.f);   // out_adj held A3
    float m = (r == c) ? 0.f : ((sel[r] && sel[c]) ? 1.f : 0.f);
    out_adj[i] = v * m;
}

// ------------------------------ orchestration ------------------------------

extern "C" void kernel_launch(void* const* d_in, const int* in_sizes, int n_in,
                              void* d_out, int out_size, void* d_ws, size_t ws_size,
                              hipStream_t stream) {
    const float* x   = (const float*)d_in[0];
    const int*   ei  = (const int*)d_in[1];
    const int*   bat = (const int*)d_in[2];
    const float *w11 = (const float*)d_in[3],  *b11 = (const float*)d_in[4];
    const float *g1  = (const float*)d_in[5],  *be1 = (const float*)d_in[6];
    const float *w12 = (const float*)d_in[7],  *b12 = (const float*)d_in[8];
    const float *w21 = (const float*)d_in[9],  *b21 = (const float*)d_in[10];
    const float *g2  = (const float*)d_in[11], *be2 = (const float*)d_in[12];
    const float *w22 = (const float*)d_in[13], *b22 = (const float*)d_in[14];
    const float *w31 = (const float*)d_in[15], *b31 = (const float*)d_in[16];
    const float *g3  = (const float*)d_in[17], *be3 = (const float*)d_in[18];
    const float *w32 = (const float*)d_in[19], *b32 = (const float*)d_in[20];

    // workspace layout (fp32 units unless noted)
    float* W     = (float*)d_ws;
    float* A     = W;                      // N*N
    float* A2    = A + kNN;                // N*N
    float* Bcat  = A2 + kNN;               // N*2D
    float* AxAxx = Bcat + kN * 2 * kD;     // N*2D
    float* hA    = AxAxx + kN * 2 * kD;    // N*HID
    float* hB    = hA + kN * kHid;         // N*HID
    float* hC    = hB + kN * kHid;         // N*HID
    float* entv  = hC + kN * kHid;         // N
    float* Vv    = entv + kN;              // N
    float* deg   = Vv + kN;                // N
    float* pv    = deg + kN;               // N
    float* dummy = pv + kN;                // N
    float* qv    = dummy + kN;             // N
    float* agg1  = qv + kN;                // N
    float* scal  = agg1 + kN;              // 16
    float* segmx = scal + 16;              // G
    float* segsm = segmx + kG;             // G
    float* mu    = segsm + kG;             // HID
    float* var   = mu + kHid;              // HID
    int*   order = (int*)(var + kHid);     // N
    int*   sel   = order + kN;             // N
    int*   rej   = sel + kN;               // N
    __bf16* Abf  = (__bf16*)(rej + kN);    // N*N bf16
    __bf16* A2bf = Abf + kNN;              // N*N bf16

    float* out_x    = (float*)d_out;       // N*D
    float* out_adj  = out_x + kN * kD;     // N*N
    float* out_sel  = out_adj + kNN;       // N
    float* out_bp   = out_sel + kN;        // N
    float* out_loss = out_bp + kN;         // 1

    const int T = 256;
    auto blk = [](int n, int t) { return (n + t - 1) / t; };

    // 0) init
    k_zero<<<blk(kNN, T), T, 0, stream>>>(A, kNN);
    k_zero<<<1, 16, 0, stream>>>(scal, 16);
    k_zeroi<<<blk(kN, T), T, 0, stream>>>(sel, kN);
    k_zeroi<<<blk(kN, T), T, 0, stream>>>(rej, kN);

    // 1) adjacency, degree, B = [x, x*x]
    k_scatter_edges<<<blk(2 * kE2, T), T, 0, stream>>>(ei, A);
    k_rowsum<<<kN, T, 0, stream>>>(A, deg);
    k_make_bcat<<<blk(kN * kD, T), T, 0, stream>>>(x, Bcat);

    // 2) [Ax | A(x*x)] via f32 WMMA (exact)
    k_gemm_f32<<<(kN / 16) * (2 * kD / 16) / 8, T, 0, stream>>>(A, Bcat, AxAxx,
                                                                kN, kN, 2 * kD);

    // 3) entropy per node
    k_compute_V<<<kN, T, 0, stream>>>(x, AxAxx, deg, Vv);
    k_segmax<<<kG, T, 0, stream>>>(Vv, bat, segmx);
    k_segsumexp<<<kG, T, 0, stream>>>(Vv, bat, segmx, segsm);
    k_ent<<<blk(kN, T), T, 0, stream>>>(Vv, bat, segmx, segsm, entv, &scal[0]);

    // 4) GIN layer 1 (in-dim 1)
    k_gemv<<<kN, T, 0, stream>>>(A, entv, agg1, kN);
    k_add_inplace<<<blk(kN, T), T, 0, stream>>>(agg1, entv, kN);
    k_outer_lin1<<<blk(kN * kHid, T), T, 0, stream>>>(agg1, w11, b11, hA);
    k_bnstats<<<kHid, T, 0, stream>>>(hA, mu, var);
    k_bnapply<<<blk(kN * kHid, T), T, 0, stream>>>(hA, mu, var, g1, be1);
    k_gemm_f32<<<(kN / 16) * (kHid / 16) / 8, T, 0, stream>>>(hA, w12, hB,
                                                              kN, kHid, kHid);
    k_bias<<<blk(kN * kHid, T), T, 0, stream>>>(hB, b12, kN * kHid, kHid);

    // 5) GIN layer 2
    k_gemm_f32<<<(kN / 16) * (kHid / 16) / 8, T, 0, stream>>>(A, hB, hC, kN, kN, kHid);
    k_add_inplace<<<blk(kN * kHid, T), T, 0, stream>>>(hC, hB, kN * kHid);
    k_gemm_f32<<<(kN / 16) * (kHid / 16) / 8, T, 0, stream>>>(hC, w21, hA,
                                                              kN, kHid, kHid);
    k_bias_relu<<<blk(kN * kHid, T), T, 0, stream>>>(hA, b21, kN * kHid, kHid);
    k_bnstats<<<kHid, T, 0, stream>>>(hA, mu, var);
    k_bnapply<<<blk(kN * kHid, T), T, 0, stream>>>(hA, mu, var, g2, be2);
    k_gemm_f32<<<(kN / 16) * (kHid / 16) / 8, T, 0, stream>>>(hA, w22, hB,
                                                              kN, kHid, kHid);
    k_bias<<<blk(kN * kHid, T), T, 0, stream>>>(hB, b22, kN * kHid, kHid);

    // 6) GIN layer 3 -> p
    k_gemm_f32<<<(kN / 16) * (kHid / 16) / 8, T, 0, stream>>>(A, hB, hC, kN, kN, kHid);
    k_add_inplace<<<blk(kN * kHid, T), T, 0, stream>>>(hC, hB, kN * kHid);
    k_gemm_f32<<<(kN / 16) * (kHid / 16) / 8, T, 0, stream>>>(hC, w31, hA,
                                                              kN, kHid, kHid);
    k_bias_relu<<<blk(kN * kHid, T), T, 0, stream>>>(hA, b31, kN * kHid, kHid);
    k_bnstats<<<kHid, T, 0, stream>>>(hA, mu, var);
    k_bnapply<<<blk(kN * kHid, T), T, 0, stream>>>(hA, mu, var, g3, be3);
    k_gemv<<<kN, T, 0, stream>>>(hA, w32, Vv, kHid);       // reuse Vv for logits
    k_sigmoid_bias<<<blk(kN, T), T, 0, stream>>>(Vv, b32, pv);

    // 7) loss + greedy state init
    k_copy<<<blk(kN, T), T, 0, stream>>>(pv, dummy, kN);
    k_gemv<<<kN, T, 0, stream>>>(A, pv, qv, kN);           // q = A @ p
    k_dot<<<blk(kN, T), T, 0, stream>>>(entv, pv, &scal[2], kN);
    k_dot<<<blk(kN, T), T, 0, stream>>>(pv, qv, &scal[3], kN);
    k_finalize_loss<<<1, 1, 0, stream>>>(scal, out_loss);

    // 8) greedy conditional expectation (sequential, single WGP)
    k_sort<<<1, 1024, 0, stream>>>(pv, order);
    k_greedy<<<1, 1024, 0, stream>>>(A, entv, order, scal, dummy, qv, sel, rej);

    // 9) A2 = A@A, A3 = A2@A via bf16 WMMA + TDM (exact: small-int operands)
    k_f2bf<<<blk(kNN, T), T, 0, stream>>>(A, Abf, kNN);
    {
        dim3 g(kN / 16, kN / 128);
        k_gemm_bf16<<<g, T, 0, stream>>>(Abf, Abf, A2, kN, kN, kN);
        k_f2bf<<<blk(kNN, T), T, 0, stream>>>(A2, A2bf, kNN);
        k_gemm_bf16<<<g, T, 0, stream>>>(A2bf, Abf, out_adj, kN, kN, kN); // A3
    }

    // 10) outputs
    k_adj_final<<<blk(kNN, T), T, 0, stream>>>(A2, out_adj, sel);
    k_outputs_node<<<blk(kN * kD, T), T, 0, stream>>>(x, sel, bat,
                                                      out_x, out_sel, out_bp);
}
